// MoELayer_14525579395103
// MI455X (gfx1250) — compile-verified, matrix operand order checked by reading, not run on hardware
//
#include <hip/hip_runtime.h>

// MoE top-2 layer for MI455X (gfx1250), wave32 + WMMA bf16 + async LDS copies.
// N=16384 tokens, D=1024, O=1024, E=8 experts, K=2.

#define NTOK 16384
#define DDIM 1024
#define ODIM 1024
#define NEXP 8

#define BM 128
#define BN 128
#define BK 32
#define KT (DDIM / BK)   // 32 K-steps
#define LDA (BK + 8)     // ushort stride -> 80 bytes/row, keeps 16B alignment

typedef __attribute__((ext_vector_type(16))) __bf16 v16bf;
typedef __attribute__((ext_vector_type(2)))  __bf16 v2bf;
typedef __attribute__((ext_vector_type(8)))  float  v8f;
typedef __attribute__((ext_vector_type(4)))  unsigned int v4u;

// Types matching __builtin_amdgcn_global_load_async_to_lds_b128's parameters:
// (v4i addrspace(1)*, v4i addrspace(3)*, imm offset, imm cpol)
typedef int v4i __attribute__((vector_size(16)));
typedef __attribute__((address_space(1))) v4i* gptr_b128;
typedef __attribute__((address_space(3))) v4i* lptr_b128;

#if __has_builtin(__builtin_amdgcn_global_load_async_to_lds_b128)
#define ASYNC_STAGE 1
#else
#define ASYNC_STAGE 0
#endif

union FragU { v4u u[2]; v16bf v; };

// Pack two f32 -> packed bf16 pair (RNE). Only used in memory-bound
// pre-convert / fallback paths.
__device__ __forceinline__ unsigned int pack_bf16(float a, float b) {
#if __has_builtin(__builtin_amdgcn_cvt_pk_bf16_f32)
    v2bf r = __builtin_amdgcn_cvt_pk_bf16_f32(a, b);
    return __builtin_bit_cast(unsigned int, r);
#else
    unsigned int ua = __float_as_uint(a);
    ua += 0x7FFFu + ((ua >> 16) & 1u);
    unsigned int ub = __float_as_uint(b);
    ub += 0x7FFFu + ((ub >> 16) & 1u);
    return (ub & 0xFFFF0000u) | (ua >> 16);
#endif
}

__device__ __forceinline__ void wait_async_zero() {
#if ASYNC_STAGE
#if __has_builtin(__builtin_amdgcn_s_wait_asynccnt)
    __builtin_amdgcn_s_wait_asynccnt(0);
#else
    asm volatile("s_wait_asynccnt 0x0" ::: "memory");
#endif
#endif
}

// ---------------------------------------------------------------------------
// Kernel 0: zero output (harness poisons it) + per-expert counters
// ---------------------------------------------------------------------------
__global__ void moe_zero_kernel(float* __restrict__ out, int* __restrict__ counts) {
    size_t i = ((size_t)blockIdx.x * blockDim.x + threadIdx.x) * 4;
    *(float4*)(out + i) = make_float4(0.f, 0.f, 0.f, 0.f);
    if (blockIdx.x == 0 && threadIdx.x < NEXP) counts[threadIdx.x] = 0;
}

// ---------------------------------------------------------------------------
// Pre-convert: x fp32 [N,D] -> bf16 [N,D]; memory bound, 8 elems/thread
// ---------------------------------------------------------------------------
__global__ __launch_bounds__(256)
void moe_conv_x_kernel(const float* __restrict__ x, unsigned short* __restrict__ xbf) {
    size_t i = ((size_t)blockIdx.x * 256 + threadIdx.x) * 8;
    const float4* s = (const float4*)(x + i);
    float4 a = s[0], b = s[1];
    uint4 r;
    r.x = pack_bf16(a.x, a.y); r.y = pack_bf16(a.z, a.w);
    r.z = pack_bf16(b.x, b.y); r.w = pack_bf16(b.z, b.w);
    *(uint4*)(xbf + i) = r;
}

// ---------------------------------------------------------------------------
// Pre-convert + transpose: expert_w fp32 [E,D,O] -> bf16 [E,O,D].
// 32x32 tile per block via LDS so both global sides stay coalesced.
// ---------------------------------------------------------------------------
__global__ __launch_bounds__(256)
void moe_conv_w_kernel(const float* __restrict__ w, unsigned short* __restrict__ wtb) {
    __shared__ unsigned short sT[32][36];          // +4 pad, rows 72B (8B-aligned)
    const int bid = blockIdx.x;
    const int e = bid >> 10;                       // 1024 tiles per expert
    const int t = bid & 1023;
    const int dBase = (t >> 5) * 32;
    const int oBase = (t & 31) * 32;
    const int tid = threadIdx.x;

    {   // read 32(d) x 32(o), coalesced along o; convert; stage
        int d = tid >> 3, o4 = (tid & 7) * 4;
        float4 v = *(const float4*)(w + (size_t)e * DDIM * ODIM +
                                    (size_t)(dBase + d) * ODIM + oBase + o4);
        *(uint2*)&sT[d][o4] = make_uint2(pack_bf16(v.x, v.y), pack_bf16(v.z, v.w));
    }
    __syncthreads();
    {   // write transposed, coalesced along d
        int o = tid >> 3, d4 = (tid & 7) * 4;
        unsigned short u0 = sT[d4 + 0][o], u1 = sT[d4 + 1][o];
        unsigned short u2 = sT[d4 + 2][o], u3 = sT[d4 + 3][o];
        uint2 r = make_uint2(((unsigned)u1 << 16) | u0, ((unsigned)u3 << 16) | u2);
        *(uint2*)(wtb + (size_t)e * ODIM * DDIM +
                  (size_t)(oBase + o) * DDIM + dBase + d4) = r;
    }
}

// ---------------------------------------------------------------------------
// Kernel 1: gate in full fp32 (top-k is order sensitive -> no bf16 here).
// One wave32 per token; butterfly reduce; append (token, prob) to buckets.
// ---------------------------------------------------------------------------
__global__ __launch_bounds__(256)
void moe_gate_kernel(const float* __restrict__ x,
                     const float* __restrict__ gate_w,
                     const float* __restrict__ gate_b,
                     int*   __restrict__ counts,
                     int*   __restrict__ toks,
                     float* __restrict__ wgts) {
    const int wave = threadIdx.x >> 5;
    const int lane = threadIdx.x & 31;
    const int n = blockIdx.x * 8 + wave;

    float acc[NEXP];
#pragma unroll
    for (int e = 0; e < NEXP; ++e) acc[e] = 0.f;

    const float* xr = x + (size_t)n * DDIM;
    for (int d = lane; d < DDIM; d += 32) {
        float xv = xr[d];
        const float4* gw = (const float4*)(gate_w + (size_t)d * NEXP);
        float4 g0 = gw[0], g1 = gw[1];
        acc[0] += xv * g0.x; acc[1] += xv * g0.y;
        acc[2] += xv * g0.z; acc[3] += xv * g0.w;
        acc[4] += xv * g1.x; acc[5] += xv * g1.y;
        acc[6] += xv * g1.z; acc[7] += xv * g1.w;
    }
#pragma unroll
    for (int m = 16; m >= 1; m >>= 1) {
#pragma unroll
        for (int e = 0; e < NEXP; ++e)
            acc[e] += __shfl_xor(acc[e], m, 32);
    }

    if (lane == 0) {
        float mx = -3.0e38f;
#pragma unroll
        for (int e = 0; e < NEXP; ++e) {
            acc[e] += gate_b[e];
            mx = fmaxf(mx, acc[e]);
        }
        float p[NEXP], sum = 0.f;
#pragma unroll
        for (int e = 0; e < NEXP; ++e) { p[e] = __expf(acc[e] - mx); sum += p[e]; }
        float inv = 1.f / sum;

        // top-1 then top-2; strict '>' keeps lowest index on ties (matches top_k)
        int e0 = 0; float p0 = p[0];
#pragma unroll
        for (int e = 1; e < NEXP; ++e) if (p[e] > p0) { p0 = p[e]; e0 = e; }
        int e1 = -1; float p1 = -1.f;
#pragma unroll
        for (int e = 0; e < NEXP; ++e) if (e != e0 && p[e] > p1) { p1 = p[e]; e1 = e; }

        int s0 = atomicAdd(&counts[e0], 1);
        toks[e0 * NTOK + s0] = n;
        wgts[e0 * NTOK + s0] = p0 * inv;   // ORIGINAL softmax prob (not renormalized)
        int s1 = atomicAdd(&counts[e1], 1);
        toks[e1 * NTOK + s1] = n;
        wgts[e1 * NTOK + s1] = p1 * inv;
    }
}

// ---------------------------------------------------------------------------
// Kernel 2: grouped expert GEMM, bf16 WMMA 16x16x32, f32 accumulate.
// Block = (expert, 128-token tile, 128-output tile), 256 threads = 8 waves.
// Double-buffered LDS, K-loop unrolled 2x (= ping-pong period).
// PRE=true + async: staging is GLOBAL_LOAD_ASYNC_TO_LDS_B128 (no staging
//   VGPRs at all; ASYNCcnt-tracked, overlaps the WMMAs).
// PRE=true no-async: register-relay b128 copies.
// PRE=false: fused fp32->bf16 conversion fallback (small workspace).
// ---------------------------------------------------------------------------
template <bool PRE>
__global__ __launch_bounds__(256, 1)
void moe_gemm_kernel(const float* __restrict__ x,
                     const unsigned short* __restrict__ xbf,
                     const float* __restrict__ expert_w,
                     const unsigned short* __restrict__ wtb,
                     const float* __restrict__ expert_b,
                     const int*   __restrict__ counts,
                     const int*   __restrict__ toks,
                     const float* __restrict__ wgts,
                     float* __restrict__ out) {
    __shared__ unsigned short sA[2][BM][LDA];   // gathered x tile, bf16
    __shared__ unsigned short sB[2][BN][LDA];   // expert_w tile, o-major
    __shared__ int   sTok[BM];
    __shared__ float sWgt[BM];

    const int mtiles = NTOK / BM;            // 128
    const int e  = blockIdx.x / mtiles;
    const int mt = blockIdx.x % mtiles;
    const int cnt = counts[e];
    const int mBase = mt * BM;
    if (mBase >= cnt) return;                // empty tile for this expert
    const int oBase = blockIdx.y * BN;

    const int tid = threadIdx.x;
    if (tid < BM) {
        int idx = mBase + tid;
        bool v = idx < cnt;
        sTok[tid] = v ? toks[e * NTOK + idx] : -1;
        sWgt[tid] = v ? wgts[e * NTOK + idx] : 0.f;
    }
    __syncthreads();

    const int lane = tid & 31;
    const int wv   = tid >> 5;
    const int wm   = wv >> 1;                // 0..3 -> row group of 32
    const int wn   = wv & 1;                 // 0..1 -> col group of 64
    const int half = lane >> 4;              // lane group (K split per WMMA layout)
    const int lr   = lane & 15;

    v8f acc[2][4] = {};                      // 64 VGPRs of f32 accumulators

    // ---- staging state (each instantiation keeps only its own set) --------
    // PRE path: 2 A-chunks + 2 B-chunks of 16B per thread per K-step.
    const int cRow0 = tid >> 2;              // 0..63 (second chunk: +64)
    const int cCol  = (tid & 3) * 8;         // ushort offset 0,8,16,24
    const unsigned short* aPtrP[2];
    const unsigned short* bPtrP[2];
    uint4 qa[2], qb[2];                      // only used when !ASYNC_STAGE
    // Fused path: fp32 loads + in-register transpose + convert.
    const int aRow0 = tid >> 3;
    const int aCol  = (tid & 7) * 4;
    const int bKg   = (tid >> 5) * 4;
    const int bOg   = (tid & 31) * 4;
    const float* aPtrF[4];
    float4 pa[4], pb[4];

    const size_t wOff = (size_t)e * DDIM * ODIM;

    // Branchless gather: invalid rows (tok<0) read token 0 (finite garbage);
    // the epilogue never stores rows with tok<0, so values are harmless.
    if constexpr (PRE) {
        int t0 = sTok[cRow0];
        int t1 = sTok[cRow0 + 64];
        aPtrP[0] = xbf + (size_t)(t0 < 0 ? 0 : t0) * DDIM + cCol;
        aPtrP[1] = xbf + (size_t)(t1 < 0 ? 0 : t1) * DDIM + cCol;
        const unsigned short* wb = wtb + (size_t)e * ODIM * DDIM;
        bPtrP[0] = wb + (size_t)(oBase + cRow0) * DDIM + cCol;
        bPtrP[1] = wb + (size_t)(oBase + cRow0 + 64) * DDIM + cCol;
    } else {
#pragma unroll
        for (int p = 0; p < 4; ++p) {
            int tok = sTok[aRow0 + p * 32];
            aPtrF[p] = x + (size_t)(tok < 0 ? 0 : tok) * DDIM + aCol;
        }
    }

    // ---- PRE + async: VGPR-free DMA of one K-step's tiles into buffer -----
    auto stage_async = [&](int kBase, int buf) {
#if ASYNC_STAGE
        __builtin_amdgcn_global_load_async_to_lds_b128(
            (gptr_b128)(aPtrP[0] + kBase), (lptr_b128)&sA[buf][cRow0][cCol], 0, 0);
        __builtin_amdgcn_global_load_async_to_lds_b128(
            (gptr_b128)(aPtrP[1] + kBase), (lptr_b128)&sA[buf][cRow0 + 64][cCol], 0, 0);
        __builtin_amdgcn_global_load_async_to_lds_b128(
            (gptr_b128)(bPtrP[0] + kBase), (lptr_b128)&sB[buf][cRow0][cCol], 0, 0);
        __builtin_amdgcn_global_load_async_to_lds_b128(
            (gptr_b128)(bPtrP[1] + kBase), (lptr_b128)&sB[buf][cRow0 + 64][cCol], 0, 0);
#endif
    };
    // ---- register-relay staging (PRE without async, and fused path) -------
    auto load_regs = [&](int kBase) {
        if constexpr (PRE) {
            qa[0] = *(const uint4*)(aPtrP[0] + kBase);
            qa[1] = *(const uint4*)(aPtrP[1] + kBase);
            qb[0] = *(const uint4*)(bPtrP[0] + kBase);
            qb[1] = *(const uint4*)(bPtrP[1] + kBase);
        } else {
#pragma unroll
            for (int p = 0; p < 4; ++p)
                pa[p] = *(const float4*)(aPtrF[p] + kBase);
#pragma unroll
            for (int i = 0; i < 4; ++i)
                pb[i] = *(const float4*)(expert_w + wOff +
                                         (size_t)(kBase + bKg + i) * ODIM + oBase + bOg);
        }
    };
    auto store_lds = [&](int buf) {
        if constexpr (PRE) {
            *(uint4*)&sA[buf][cRow0][cCol]      = qa[0];
            *(uint4*)&sA[buf][cRow0 + 64][cCol] = qa[1];
            *(uint4*)&sB[buf][cRow0][cCol]      = qb[0];
            *(uint4*)&sB[buf][cRow0 + 64][cCol] = qb[1];
        } else {
#pragma unroll
            for (int p = 0; p < 4; ++p)
                *(uint2*)&sA[buf][aRow0 + p * 32][aCol] =
                    make_uint2(pack_bf16(pa[p].x, pa[p].y), pack_bf16(pa[p].z, pa[p].w));
            const float* f0 = (const float*)&pb[0];
            const float* f1 = (const float*)&pb[1];
            const float* f2 = (const float*)&pb[2];
            const float* f3 = (const float*)&pb[3];
#pragma unroll
            for (int i = 0; i < 4; ++i)
                *(uint2*)&sB[buf][bOg + i][bKg] =
                    make_uint2(pack_bf16(f0[i], f1[i]), pack_bf16(f2[i], f3[i]));
        }
    };

    constexpr bool USE_ASYNC = PRE && (ASYNC_STAGE != 0);

    // prologue: stage K-step 0 into buffer 0
    if constexpr (USE_ASYNC) {
        stage_async(0, 0);
        wait_async_zero();
    } else {
        load_regs(0);
        store_lds(0);
    }
    __syncthreads();

    // Unroll exactly 2 (ping-pong period): `cur` is compile-time 0/1, LDS
    // offsets are immediates, and register liveness stays bounded.
#pragma unroll 2
    for (int ks = 0; ks < KT; ++ks) {
        const int cur = ks & 1;
        const bool pref = (ks + 1) < KT;
        if (pref) {
            if constexpr (USE_ASYNC) stage_async((ks + 1) * BK, 1 - cur);
            else                     load_regs((ks + 1) * BK);
        }

        // Fragment loads per documented 16-bit A layout:
        // lanes 0-15: K 0-7 (V0-3) + K 16-23 (V4-7); lanes 16-31: K 8-15 + 24-31
        FragU a[2], b[4];
#pragma unroll
        for (int i = 0; i < 2; ++i) {
            int r = wm * 32 + i * 16 + lr;
            a[i].u[0] = *(const v4u*)&sA[cur][r][half * 8];
            a[i].u[1] = *(const v4u*)&sA[cur][r][16 + half * 8];
        }
#pragma unroll
        for (int j = 0; j < 4; ++j) {
            int c = wn * 64 + j * 16 + lr;
            b[j].u[0] = *(const v4u*)&sB[cur][c][half * 8];
            b[j].u[1] = *(const v4u*)&sB[cur][c][16 + half * 8];
        }
#pragma unroll
        for (int i = 0; i < 2; ++i)
#pragma unroll
            for (int j = 0; j < 4; ++j)
                acc[i][j] = __builtin_amdgcn_wmma_f32_16x16x32_bf16(
                    false, a[i].v, false, b[j].v, (short)0, acc[i][j], false, false);

        if (pref) {
            if constexpr (USE_ASYNC) wait_async_zero();   // per-wave; barrier covers block
            else                     store_lds(1 - cur);
        }
        __syncthreads();
    }

    // Epilogue: out[tok, o] += weight * (gemm + bias). Exactly 2 commutative
    // fp32 atomic adds per output element -> deterministic.
    float bias[4];
#pragma unroll
    for (int j = 0; j < 4; ++j)
        bias[j] = expert_b[e * ODIM + oBase + wn * 64 + j * 16 + lr];

#pragma unroll
    for (int i = 0; i < 2; ++i) {
        int mRow = wm * 32 + i * 16 + half * 8;   // C layout: m = r + half*8
#pragma unroll
        for (int r = 0; r < 8; ++r) {
            int m = mRow + r;
            int tok = sTok[m];
            if (tok < 0) continue;
            float w = sWgt[m];
            float* orow = out + (size_t)tok * ODIM + oBase + wn * 64 + lr;
#pragma unroll
            for (int j = 0; j < 4; ++j)
                atomicAdd(orow + j * 16, w * (acc[i][j][r] + bias[j]));
        }
    }
}

// ---------------------------------------------------------------------------
extern "C" void kernel_launch(void* const* d_in, const int* in_sizes, int n_in,
                              void* d_out, int out_size, void* d_ws, size_t ws_size,
                              hipStream_t stream) {
    const float* x        = (const float*)d_in[0];
    const float* gate_w   = (const float*)d_in[1];
    const float* gate_b   = (const float*)d_in[2];
    const float* expert_w = (const float*)d_in[3];
    const float* expert_b = (const float*)d_in[4];
    float* out = (float*)d_out;

    // ws layout: [counts pad 256B][toks E*N i32][wgts E*N f32][xbf][wtb]
    const size_t off_toks = 256;
    const size_t off_wgts = off_toks + (size_t)NEXP * NTOK * sizeof(int);
    size_t off_xbf = off_wgts + (size_t)NEXP * NTOK * sizeof(float);
    off_xbf = (off_xbf + 255) & ~(size_t)255;
    const size_t off_wtb = off_xbf + (size_t)NTOK * DDIM * 2;
    const size_t need    = off_wtb + (size_t)NEXP * ODIM * DDIM * 2;

    int*   counts = (int*)d_ws;
    int*   toks   = (int*)((char*)d_ws + off_toks);
    float* wgts   = (float*)((char*)d_ws + off_wgts);
    unsigned short* xbf = (unsigned short*)((char*)d_ws + off_xbf);
    unsigned short* wtb = (unsigned short*)((char*)d_ws + off_wtb);

    const bool pre = (ws_size >= need);      // constant across calls -> deterministic

    moe_zero_kernel<<<(NTOK * ODIM) / (256 * 4), 256, 0, stream>>>(out, counts);
    moe_gate_kernel<<<NTOK / 8, 256, 0, stream>>>(x, gate_w, gate_b, counts, toks, wgts);

    dim3 grid(NEXP * (NTOK / BM), ODIM / BN);   // (1024, 8); ~3/4 early-exit
    if (pre) {
        moe_conv_x_kernel<<<(NTOK * DDIM) / (256 * 8), 256, 0, stream>>>(x, xbf);
        moe_conv_w_kernel<<<NEXP * (DDIM / 32) * (ODIM / 32), 256, 0, stream>>>(expert_w, wtb);
        moe_gemm_kernel<true><<<grid, 256, 0, stream>>>(x, xbf, expert_w, wtb, expert_b,
                                                        counts, toks, wgts, out);
    } else {
        moe_gemm_kernel<false><<<grid, 256, 0, stream>>>(x, xbf, expert_w, wtb, expert_b,
                                                         counts, toks, wgts, out);
    }
}
